// GCNWithMultiHeadGATAndTCN_28956669509711
// MI455X (gfx1250) — matrix operand order repaired and checked by reading, not compiled, not run on hardware
//
#include <hip/hip_runtime.h>

typedef __bf16 bf16;
typedef __attribute__((ext_vector_type(16))) __bf16 v16bf;
typedef __attribute__((ext_vector_type(8)))  __bf16 v8bf;
typedef __attribute__((ext_vector_type(8)))  float  v8f;

#define B_   16
#define N_   1024
#define NF_  128
#define NH_  256
#define NO_  256
#define NHD_ 4
#define DH_  64
#define ROWS_ (B_ * N_)            // 16384 flat (b,n) rows
#define EPS_  1e-5f

// ---------------------------------------------------------------- helpers

__device__ __forceinline__ float clamp01(float a) { return fminf(fmaxf(a, 0.f), 1.f); }

__device__ __forceinline__ v8f wmma_bf16(v16bf a, v16bf b, v8f c) {
  // D = A(16x32 bf16) * B(32x16 bf16) + C(16x16 f32)
  return __builtin_amdgcn_wmma_f32_16x16x32_bf16(false, a, false, b, (short)0, c, false, false);
}

__device__ __forceinline__ v16bf zero_af() {
  v16bf a;
#pragma unroll
  for (int e = 0; e < 16; ++e) a[e] = (__bf16)0.0f;
  return a;
}
__device__ __forceinline__ v8f zero_cf() {
  v8f c;
#pragma unroll
  for (int v = 0; v < 8; ++v) c[v] = 0.0f;
  return c;
}

// A fragment 16x32 bf16. rowp already points at &A[row_of_this_lane][k0].
// Lane layout (wave32): lanes 0-15 / 16-31 share rows, half selects K sub-block:
//   e<8  -> k = half*8 + e ; e>=8 -> k = 16 + half*8 + (e-8)
__device__ __forceinline__ v16bf load_a_rowptr(const bf16* rowp) {
  int half = (threadIdx.x >> 4) & 1;
  const v8bf* p = (const v8bf*)(rowp + half * 8);
  v8bf lo = p[0];
  v8bf hi = p[2];                 // +16 elements
  v16bf r;
#pragma unroll
  for (int e = 0; e < 8; ++e) { r[e] = lo[e]; r[8 + e] = hi[e]; }
  return r;
}
__device__ __forceinline__ v16bf load_a16x32(const bf16* tile, int ld) {
  int row = threadIdx.x & 15;
  return load_a_rowptr(tile + (size_t)row * ld);
}

// B fragment 32x16 bf16 read from B^T stored row-major [ncol][k]:
//   lane col n = lane&15, e<8 -> k = half*8 + e ; e>=8 -> k = 16 + half*8 + (e-8)
__device__ __forceinline__ v16bf load_b32x16(const bf16* tileT, int ld) {
  int n = threadIdx.x & 15, half = (threadIdx.x >> 4) & 1;
  const v8bf* p = (const v8bf*)(tileT + (size_t)n * ld + half * 8);
  v8bf lo = p[0], hi = p[2];
  v16bf r;
#pragma unroll
  for (int e = 0; e < 8; ++e) { r[e] = lo[e]; r[8 + e] = hi[e]; }
  return r;
}

// ---------------------------------------------------------------- prep kernels

__global__ void k_cvt_bf16(const float* __restrict__ s, bf16* __restrict__ d, int n) {
  for (int i = blockIdx.x * blockDim.x + threadIdx.x; i < n; i += gridDim.x * blockDim.x)
    d[i] = (bf16)s[i];
}

// Wcat^T [512][128]: rows 0..255 = W1^T, rows 256..511 = Wres^T  (W is [128][256])
__global__ void k_pack_wcat(const float* __restrict__ W1, const float* __restrict__ Wres,
                            bf16* __restrict__ wcat) {
  int i = blockIdx.x * blockDim.x + threadIdx.x;       // 512*128
  int c = i >> 7, f = i & 127;
  float v = (c < NH_) ? W1[f * NH_ + c] : Wres[f * NH_ + (c - NH_)];
  wcat[i] = (bf16)v;
}

// Wh^T flat [256][256]: row c = k*64+d, col h ; Wh is (4,256,64)
__global__ void k_pack_wh(const float* __restrict__ Wh, bf16* __restrict__ whT) {
  int i = blockIdx.x * blockDim.x + threadIdx.x;       // 256*256
  int c = i >> 8, hh = i & 255;
  int k = c >> 6, d = c & 63;
  whT[i] = (bf16)Wh[((size_t)k * NH_ + hh) * DH_ + d];
}

// Wc^T [256][768]: row c_out, col k = kk*256 + ci ; conv_w is (256,256,3)
__global__ void k_pack_wc(const float* __restrict__ cw, bf16* __restrict__ wcT) {
  int i = blockIdx.x * blockDim.x + threadIdx.x;       // 256*768
  int co = i / 768, kx = i % 768;
  int kk = kx >> 8, ci = kx & 255;
  wcT[i] = (bf16)cw[((size_t)co * NO_ + ci) * 3 + kk];
}

// ---------------------------------------------------------------- GEMM 1: x@[W1|Wres]

// M=16384, K=128, N=512; 2x2 register blocking (32x32 per wave).
// cols<256 -> supportT (bf16, transposed), cols>=256 -> residual (f32)
__global__ void k_gemm1(const bf16* __restrict__ xbf, const bf16* __restrict__ wcat,
                        bf16* __restrict__ supportT, float* __restrict__ residual) {
  int wid = (blockIdx.x * blockDim.x + threadIdx.x) >> 5;
  int mt = wid >> 4, nt = wid & 15;                    // 512 x 16 blocks of 32x32
  const bf16* A0 = xbf + (size_t)mt * 32 * NF_;
  const bf16* A1 = A0 + (size_t)16 * NF_;
  const bf16* B0 = wcat + (size_t)nt * 32 * NF_;
  const bf16* B1 = B0 + (size_t)16 * NF_;
  v8f acc[2][2];
  acc[0][0] = acc[0][1] = acc[1][0] = acc[1][1] = zero_cf();
#pragma unroll
  for (int ks = 0; ks < NF_; ks += 32) {
    v16bf a0 = load_a16x32(A0 + ks, NF_), a1 = load_a16x32(A1 + ks, NF_);
    v16bf b0 = load_b32x16(B0 + ks, NF_), b1 = load_b32x16(B1 + ks, NF_);
    acc[0][0] = wmma_bf16(a0, b0, acc[0][0]);
    acc[0][1] = wmma_bf16(a0, b1, acc[0][1]);
    acc[1][0] = wmma_bf16(a1, b0, acc[1][0]);
    acc[1][1] = wmma_bf16(a1, b1, acc[1][1]);
  }

  int ncol = threadIdx.x & 15, half = (threadIdx.x >> 4) & 1;
#pragma unroll
  for (int ti = 0; ti < 2; ++ti) {
#pragma unroll
    for (int tj = 0; tj < 2; ++tj) {
      int m0 = mt * 32 + ti * 16;
      int c0 = nt * 32 + tj * 16;
      v8f a = acc[ti][tj];
      if (c0 < NH_) {
        // supportT[b][h][m] : each lane stores 8 contiguous bf16 along m
        int b = m0 >> 10, ml = m0 & (N_ - 1);
        bf16* ct = supportT + ((size_t)b * NH_ + c0 + ncol) * N_ + ml + half * 8;
        v8bf o;
#pragma unroll
        for (int v = 0; v < 8; ++v) o[v] = (bf16)a[v];
        *(v8bf*)ct = o;
      } else {
        float* R = residual + (size_t)m0 * NH_ + (c0 - NH_);
#pragma unroll
        for (int v = 0; v < 8; ++v) R[(size_t)(v + half * 8) * NH_ + ncol] = a[v];
      }
    }
  }
}

// ---------------------------------------------------------------- GEMM 2: adj @ support

// per b: M=1024 (n), K=1024 (m), N=256 (h); 2x2 blocking. Epilogue: +b1, relu, res-weight
__global__ void k_gemm2(const bf16* __restrict__ adjbf, const bf16* __restrict__ supportT,
                        const float* __restrict__ b1, const float* __restrict__ residual,
                        const float* __restrict__ alpha_gc, float* __restrict__ hpre) {
  int wid = (blockIdx.x * blockDim.x + threadIdx.x) >> 5;
  int b = wid >> 8, mt = (wid >> 3) & 31, nt = wid & 7; // 16 x 32 x 8 blocks of 32x32
  const bf16* A0 = adjbf + (size_t)mt * 32 * N_;
  const bf16* A1 = A0 + (size_t)16 * N_;
  const bf16* B0 = supportT + ((size_t)b * NH_ + nt * 32) * N_;
  const bf16* B1 = B0 + (size_t)16 * N_;
  v8f acc[2][2];
  acc[0][0] = acc[0][1] = acc[1][0] = acc[1][1] = zero_cf();
  for (int ks = 0; ks < N_; ks += 32) {
    v16bf a0 = load_a16x32(A0 + ks, N_), a1 = load_a16x32(A1 + ks, N_);
    v16bf b0f = load_b32x16(B0 + ks, N_), b1f = load_b32x16(B1 + ks, N_);
    acc[0][0] = wmma_bf16(a0, b0f, acc[0][0]);
    acc[0][1] = wmma_bf16(a0, b1f, acc[0][1]);
    acc[1][0] = wmma_bf16(a1, b0f, acc[1][0]);
    acc[1][1] = wmma_bf16(a1, b1f, acc[1][1]);
  }

  int ncol = threadIdx.x & 15, half = (threadIdx.x >> 4) & 1;
  float al = clamp01(alpha_gc[0]);
#pragma unroll
  for (int ti = 0; ti < 2; ++ti) {
#pragma unroll
    for (int tj = 0; tj < 2; ++tj) {
      int c = nt * 32 + tj * 16 + ncol;
      float bias = b1[c];
      size_t r0 = (size_t)b * N_ + mt * 32 + ti * 16 + half * 8;
      v8f a = acc[ti][tj];
#pragma unroll
      for (int v = 0; v < 8; ++v) {
        size_t r = r0 + v;
        float gc = fmaxf(a[v] + bias, 0.f);
        float rv = residual[r * NH_ + c];
        hpre[r * NH_ + c] = al * gc + (1.f - al) * rv;
      }
    }
  }
}

// ---------------------------------------------------------------- BatchNorm (train mode)

__global__ void k_bn_stats(const float* __restrict__ v, float* __restrict__ stats,
                           int C, int S) {
  __shared__ float sh[256], sh2[256];
  int c = blockIdx.x;
  float s = 0.f, s2 = 0.f;
  for (int i = threadIdx.x; i < S; i += blockDim.x) {
    float x = v[(size_t)i * C + c];
    s += x; s2 += x * x;
  }
  sh[threadIdx.x] = s; sh2[threadIdx.x] = s2;
  __syncthreads();
  for (int o = 128; o > 0; o >>= 1) {
    if ((int)threadIdx.x < o) { sh[threadIdx.x] += sh[threadIdx.x + o]; sh2[threadIdx.x] += sh2[threadIdx.x + o]; }
    __syncthreads();
  }
  if (threadIdx.x == 0) {
    float mean = sh[0] / (float)S;
    float var = sh2[0] / (float)S - mean * mean;
    stats[c] = mean;
    stats[C + c] = rsqrtf(var + EPS_);
  }
}

__global__ void k_bn_relu_apply(const float* __restrict__ src, const float* __restrict__ gamma,
                                const float* __restrict__ beta, const float* __restrict__ stats,
                                float* __restrict__ dstF, bf16* __restrict__ dstB, int n, int C) {
  for (int i = blockIdx.x * blockDim.x + threadIdx.x; i < n; i += gridDim.x * blockDim.x) {
    int c = i & (C - 1);
    float x = (src[i] - stats[c]) * stats[C + c] * gamma[c] + beta[c];
    x = fmaxf(x, 0.f);
    dstF[i] = x;
    dstB[i] = (bf16)x;
  }
}

// ---------------------------------------------------------------- GEMM 3: head projection

// M=16384, K=256, N=256 (4 heads x 64); 2x2 blocking.
// Writes hpbf [b][k][n][d] and hpT [b][k][d][n]
__global__ void k_gemm3(const bf16* __restrict__ hbf, const bf16* __restrict__ whT,
                        bf16* __restrict__ hpbf, bf16* __restrict__ hpT) {
  int wid = (blockIdx.x * blockDim.x + threadIdx.x) >> 5;
  int mt = wid >> 3, nt = wid & 7;                     // 512 x 8 blocks of 32x32
  const bf16* A0 = hbf + (size_t)mt * 32 * NH_;
  const bf16* A1 = A0 + (size_t)16 * NH_;
  const bf16* B0 = whT + (size_t)nt * 32 * NH_;
  const bf16* B1 = B0 + (size_t)16 * NH_;
  v8f acc[2][2];
  acc[0][0] = acc[0][1] = acc[1][0] = acc[1][1] = zero_cf();
#pragma unroll
  for (int ks = 0; ks < NH_; ks += 32) {
    v16bf a0 = load_a16x32(A0 + ks, NH_), a1 = load_a16x32(A1 + ks, NH_);
    v16bf b0f = load_b32x16(B0 + ks, NH_), b1f = load_b32x16(B1 + ks, NH_);
    acc[0][0] = wmma_bf16(a0, b0f, acc[0][0]);
    acc[0][1] = wmma_bf16(a0, b1f, acc[0][1]);
    acc[1][0] = wmma_bf16(a1, b0f, acc[1][0]);
    acc[1][1] = wmma_bf16(a1, b1f, acc[1][1]);
  }

  int ncol = threadIdx.x & 15, half = (threadIdx.x >> 4) & 1;
#pragma unroll
  for (int ti = 0; ti < 2; ++ti) {
#pragma unroll
    for (int tj = 0; tj < 2; ++tj) {
      int c = nt * 32 + tj * 16 + ncol;
      int hk = c >> 6, d = c & 63;
      int m0 = mt * 32 + ti * 16;
      int b = m0 >> 10, nn0 = (m0 & (N_ - 1)) + half * 8;
      size_t bk = (size_t)b * NHD_ + hk;
      v8f a = acc[ti][tj];
      // strided bf16 stores: hpbf[b][k][n][d]
#pragma unroll
      for (int v = 0; v < 8; ++v)
        hpbf[(bk * N_ + nn0 + v) * DH_ + d] = (bf16)a[v];
      // contiguous transposed store: hpT[b][k][d][n]
      v8bf o;
#pragma unroll
      for (int v = 0; v < 8; ++v) o[v] = (bf16)a[v];
      *(v8bf*)(hpT + (bk * DH_ + d) * N_ + nn0) = o;
    }
  }
}

// ---------------------------------------------------------------- fused GAT attention (flash)

// one wave per (b, head, 16-row tile); online softmax over 1024 cols, never materializes NxN
__global__ void k_attn(const bf16* __restrict__ hpbf, const bf16* __restrict__ hpT,
                       const float* __restrict__ h, const float* __restrict__ alpha_gat,
                       float* __restrict__ h2, bf16* __restrict__ h2bf) {
  __shared__ __align__(16) bf16 smem[8][16 * 48];      // P transpose bounce, padded stride
  int wave = threadIdx.x >> 5;
  int wid = blockIdx.x * 8 + wave;
  int ncol = threadIdx.x & 15, half = (threadIdx.x >> 4) & 1;
  int b = wid >> 8, hk = (wid >> 6) & 3, i0 = (wid & 63) << 4;
  size_t bk = (size_t)b * NHD_ + hk;

  const bf16* Q = hpbf + (bk * N_ + i0) * DH_;
  v16bf aq0 = load_a16x32(Q, DH_);
  v16bf aq1 = load_a16x32(Q + 32, DH_);

  float m[8], l[8];
  v8f accv[4];
#pragma unroll
  for (int v = 0; v < 8; ++v) { m[v] = -3.0e38f; l[v] = 0.f; }
#pragma unroll
  for (int t = 0; t < 4; ++t) accv[t] = zero_cf();

  bf16* sm = smem[wave];

  for (int j0 = 0; j0 < N_; j0 += 32) {
    const bf16* Kb = hpbf + (bk * N_ + j0) * DH_;
    v8f s0 = zero_cf(), s1 = zero_cf();
    s0 = wmma_bf16(aq0, load_b32x16(Kb, DH_), s0);
    s0 = wmma_bf16(aq1, load_b32x16(Kb + 32, DH_), s0);
    const bf16* Kb2 = Kb + 16 * DH_;
    s1 = wmma_bf16(aq0, load_b32x16(Kb2, DH_), s1);
    s1 = wmma_bf16(aq1, load_b32x16(Kb2 + 32, DH_), s1);

#pragma unroll
    for (int v = 0; v < 8; ++v) {
      s0[v] = s0[v] > 0.f ? s0[v] : 0.2f * s0[v];      // leaky_relu(0.2)
      s1[v] = s1[v] > 0.f ? s1[v] : 0.2f * s1[v];
    }
    // online softmax update (rows live per-VGPR, cols across 16-lane halves)
#pragma unroll
    for (int v = 0; v < 8; ++v) {
      float rmax = fmaxf(s0[v], s1[v]);
      for (int o = 8; o >= 1; o >>= 1) rmax = fmaxf(rmax, __shfl_xor(rmax, o));
      float mn = fmaxf(m[v], rmax);
      float sc = __expf(m[v] - mn);
      m[v] = mn;
      float p0 = __expf(s0[v] - mn);
      float p1 = __expf(s1[v] - mn);
      float rs = p0 + p1;
      for (int o = 8; o >= 1; o >>= 1) rs += __shfl_xor(rs, o);
      l[v] = l[v] * sc + rs;
      s0[v] = p0; s1[v] = p1;
#pragma unroll
      for (int t = 0; t < 4; ++t) accv[t][v] *= sc;
    }
    // P : D-layout -> A-layout via per-wave LDS tile
    asm volatile("" ::: "memory");
#pragma unroll
    for (int v = 0; v < 8; ++v) {
      int r = (v + half * 8) * 48;
      sm[r + ncol] = (bf16)s0[v];
      sm[r + 16 + ncol] = (bf16)s1[v];
    }
    asm volatile("s_wait_dscnt 0x0" ::: "memory");
    v16bf pa = load_a16x32(sm, 48);
    // PV: V fragments from hpT[b][k][d][n] (already B^T layout, contiguous along j)
#pragma unroll
    for (int t = 0; t < 4; ++t) {
      const bf16* Vb = hpT + (bk * DH_ + t * 16) * N_ + j0;
      accv[t] = wmma_bf16(pa, load_b32x16(Vb, N_), accv[t]);
    }
  }

  float al = clamp01(alpha_gat[0]);
#pragma unroll
  for (int t = 0; t < 4; ++t) {
    int c = hk * DH_ + t * 16 + ncol;
#pragma unroll
    for (int v = 0; v < 8; ++v) {
      size_t n = (size_t)i0 + v + half * 8;
      float g = accv[t][v] / l[v];
      size_t idx = ((size_t)b * N_ + n) * NO_ + c;
      float o = al * g + (1.f - al) * h[idx];
      h2[idx] = o;
      h2bf[idx] = (bf16)o;
    }
  }
}

// ---------------------------------------------------------------- conv1d(k=3) as K=768 GEMM

// M=16384, K=768, N=256; 2x2 blocking with per-row boundary guards (zero padding)
__global__ void k_conv(const bf16* __restrict__ h2bf, const bf16* __restrict__ wcT,
                       const float* __restrict__ cb, float* __restrict__ y) {
  int wid = (blockIdx.x * blockDim.x + threadIdx.x) >> 5;
  int mt = wid >> 3, nt = wid & 7;                     // 512 x 8 blocks of 32x32
  int m0 = mt * 32, b = m0 >> 10, n0 = m0 & (N_ - 1);
  int row = threadIdx.x & 15, half = (threadIdx.x >> 4) & 1;
  v8f acc[2][2];
  acc[0][0] = acc[0][1] = acc[1][0] = acc[1][1] = zero_cf();
  for (int ks = 0; ks < 768; ks += 32) {
    int kk = ks >> 8, ci0 = ks & 255;                  // 256 % 32 == 0 -> kk uniform per chunk
    int ns0 = n0 + row + kk - 1;
    int ns1 = ns0 + 16;
    v16bf a0 = (ns0 >= 0 && ns0 < N_)
                   ? load_a_rowptr(h2bf + ((size_t)b * N_ + ns0) * NO_ + ci0)
                   : zero_af();
    v16bf a1 = (ns1 >= 0 && ns1 < N_)
                   ? load_a_rowptr(h2bf + ((size_t)b * N_ + ns1) * NO_ + ci0)
                   : zero_af();
    v16bf b0f = load_b32x16(wcT + (size_t)nt * 32 * 768 + ks, 768);
    v16bf b1f = load_b32x16(wcT + ((size_t)nt * 32 + 16) * 768 + ks, 768);
    acc[0][0] = wmma_bf16(a0, b0f, acc[0][0]);
    acc[0][1] = wmma_bf16(a0, b1f, acc[0][1]);
    acc[1][0] = wmma_bf16(a1, b0f, acc[1][0]);
    acc[1][1] = wmma_bf16(a1, b1f, acc[1][1]);
  }
  int ncol = threadIdx.x & 15;
#pragma unroll
  for (int ti = 0; ti < 2; ++ti) {
#pragma unroll
    for (int tj = 0; tj < 2; ++tj) {
      int c = nt * 32 + tj * 16 + ncol;
      float bias = cb[c];
      v8f a = acc[ti][tj];
#pragma unroll
      for (int v = 0; v < 8; ++v) {
        size_t r = (size_t)m0 + ti * 16 + v + half * 8;
        y[r * NO_ + c] = a[v] + bias;
      }
    }
  }
}

// ---------------------------------------------------------------- final: bn -> relu -> res-weight

__global__ void k_final(const float* __restrict__ y, const float* __restrict__ gamma,
                        const float* __restrict__ beta, const float* __restrict__ stats,
                        const float* __restrict__ alpha_tcn, const float* __restrict__ h2,
                        float* __restrict__ out, int n) {
  float al = clamp01(alpha_tcn[0]);
  for (int i = blockIdx.x * blockDim.x + threadIdx.x; i < n; i += gridDim.x * blockDim.x) {
    int c = i & (NO_ - 1);
    float x = (y[i] - stats[c]) * stats[NO_ + c] * gamma[c] + beta[c];
    x = fmaxf(x, 0.f);
    out[i] = al * x + (1.f - al) * h2[i];
  }
}

// ---------------------------------------------------------------- host launch

extern "C" void kernel_launch(void* const* d_in, const int* in_sizes, int n_in,
                              void* d_out, int out_size, void* d_ws, size_t ws_size,
                              hipStream_t stream) {
  const float* x        = (const float*)d_in[0];
  const float* adj      = (const float*)d_in[1];
  const float* W1       = (const float*)d_in[2];
  const float* b1       = (const float*)d_in[3];
  const float* Wres     = (const float*)d_in[4];
  const float* alpha_gc = (const float*)d_in[5];
  const float* g1       = (const float*)d_in[6];
  const float* be1      = (const float*)d_in[7];
  const float* Wh       = (const float*)d_in[8];
  const float* alpha_ga = (const float*)d_in[9];
  const float* cw       = (const float*)d_in[10];
  const float* cb       = (const float*)d_in[11];
  const float* g2       = (const float*)d_in[12];
  const float* be2      = (const float*)d_in[13];
  const float* alpha_tc = (const float*)d_in[14];
  float* out = (float*)d_out;

  char* ws = (char*)d_ws;
  size_t off = 0;
  auto take = [&](size_t bytes) -> char* {
    char* p = ws + off;
    off = (off + bytes + 255) & ~(size_t)255;
    return p;
  };
  bf16*  xbf      = (bf16*)take((size_t)ROWS_ * NF_ * 2);
  bf16*  adjbf    = (bf16*)take((size_t)N_ * N_ * 2);
  bf16*  wcat     = (bf16*)take((size_t)512 * NF_ * 2);
  bf16*  whT      = (bf16*)take((size_t)256 * NH_ * 2);
  bf16*  wcT      = (bf16*)take((size_t)NO_ * 768 * 2);
  bf16*  supportT = (bf16*)take((size_t)B_ * NH_ * N_ * 2);
  float* residual = (float*)take((size_t)ROWS_ * NH_ * 4);
  float* hpre     = (float*)take((size_t)ROWS_ * NH_ * 4);
  float* hF       = (float*)take((size_t)ROWS_ * NH_ * 4);
  bf16*  hbf      = (bf16*)take((size_t)ROWS_ * NH_ * 2);
  bf16*  hpbf     = (bf16*)take((size_t)B_ * NHD_ * N_ * DH_ * 2);
  bf16*  hpT      = (bf16*)take((size_t)B_ * NHD_ * DH_ * N_ * 2);
  float* h2       = (float*)take((size_t)ROWS_ * NO_ * 4);
  bf16*  h2bf     = (bf16*)take((size_t)ROWS_ * NO_ * 2);
  float* yb       = (float*)take((size_t)ROWS_ * NO_ * 4);
  float* stats1   = (float*)take(2 * NH_ * 4);
  float* stats2   = (float*)take(2 * NO_ * 4);

  // --- prep: bf16 conversions + weight transposes ---
  k_cvt_bf16<<<2048, 256, 0, stream>>>(x, xbf, ROWS_ * NF_);
  k_cvt_bf16<<<2048, 256, 0, stream>>>(adj, adjbf, N_ * N_);
  k_pack_wcat<<<(512 * NF_) / 256, 256, 0, stream>>>(W1, Wres, wcat);
  k_pack_wh<<<(256 * NH_) / 256, 256, 0, stream>>>(Wh, whT);
  k_pack_wc<<<(NO_ * 768) / 256, 256, 0, stream>>>(cw, wcT);

  // --- graph conv path ---
  k_gemm1<<<1024, 256, 0, stream>>>(xbf, wcat, supportT, residual);      // 8192 waves
  k_gemm2<<<512, 256, 0, stream>>>(adjbf, supportT, b1, residual, alpha_gc, hpre); // 4096 waves
  k_bn_stats<<<NH_, 256, 0, stream>>>(hpre, stats1, NH_, ROWS_);
  k_bn_relu_apply<<<2048, 256, 0, stream>>>(hpre, g1, be1, stats1, hF, hbf, ROWS_ * NH_, NH_);

  // --- GAT ---
  k_gemm3<<<512, 256, 0, stream>>>(hbf, whT, hpbf, hpT);                 // 4096 waves
  k_attn<<<512, 256, 0, stream>>>(hpbf, hpT, hF, alpha_ga, h2, h2bf);

  // --- TCN ---
  k_conv<<<512, 256, 0, stream>>>(h2bf, wcT, cb, yb);                    // 4096 waves
  k_bn_stats<<<NO_, 256, 0, stream>>>(yb, stats2, NO_, ROWS_);
  k_final<<<2048, 256, 0, stream>>>(yb, g2, be2, stats2, alpha_tc, h2, out, ROWS_ * NO_);
  (void)in_sizes; (void)n_in; (void)out_size; (void)ws_size;
}
